// WindowedLSTMCell_86998857548217
// MI455X (gfx1250) — compile-verified, matrix operand order checked by reading, not run on hardware
//
#include <hip/hip_runtime.h>
#include <hip/hip_bf16.h>
#include <stdint.h>

// ---------------- problem constants ----------------
#define T_STEPS 600
#define B_SZ    128
#define D_IN    3
#define U_SZ    400
#define W_SZ    80
#define L_SZ    64
#define K_MIX   10
#define OUTC    (U_SZ + W_SZ)          // 480

// packed-K layout for the fused input [x | h | w] (bf16), padded to 512
//   cols 0..2   : x_t
//   cols 8..407 : h
//   cols 416..495: w
//   everything else zero (matching weight rows are zero)
#define KPAD    512
#define XOFF_H  8
#define XOFF_W  416
#define KCH     16                      // 512/32 K-chunks for the big GEMM
#define MP_KCH  13                      // 416/32 K-chunks for mixture GEMM
#define NCOLS   1600                    // 4U
#define UTILES  25                      // 400/16 unit tiles (each -> 4 gate col-tiles)

#define NWG      25
#define NTHREADS 256
#define NWAVES   8

#define XCAT_ELEMS (B_SZ * KPAD)        // 65536 bf16 per buffer

// ---------------- workspace layout (bytes) ----------------
#define WS_CNT    0u
#define WS_WP     256u                                  // ushort[100*16*32*16] = 819200
#define WS_MP     (WS_WP + 819200u*2u)                  // ushort[2*13*32*16]  = 13312
#define WS_XCAT   (WS_MP + 13312u*2u)                   // ushort[2*65536]
#define WS_C      (WS_XCAT + 2u*XCAT_ELEMS*2u)          // float[128*400]
#define WS_ALPHA  (WS_C + 128u*400u*4u)                 // float[128*10]
#define WS_BETA   (WS_ALPHA + 1280u*4u)
#define WS_KAPPA  (WS_BETA + 1280u*4u)

// ---------------- types ----------------
typedef __attribute__((ext_vector_type(4)))  unsigned int v4u;
typedef __attribute__((ext_vector_type(8)))  float        v8f;
typedef __bf16 bf16_t;
typedef __attribute__((ext_vector_type(16))) bf16_t       v16bf;

union Frag { v4u u[2]; v16bf h; };

__device__ __forceinline__ unsigned short f2bf(float f) {
    unsigned u = __float_as_uint(f);
    unsigned r = u + 0x7FFFu + ((u >> 16) & 1u);   // round-to-nearest-even
    return (unsigned short)(r >> 16);
}

// per-lane K index inside a 32-wide K chunk for the documented 16-bit A layout
// (lane<16: K = {0..7,16..23}; lane>=16: K = {8..15,24..31})
__device__ __forceinline__ int klocal_of(int lane, int j) {
    int half = lane >> 4;
    return (j < 8) ? (half * 8 + j) : (16 + half * 8 + (j - 8));
}

__device__ __forceinline__ float sigmoidf_(float x) {
    return __builtin_amdgcn_rcpf(1.0f + __expf(-x));
}
__device__ __forceinline__ float tanhf_(float x) {
    return 1.0f - 2.0f * __builtin_amdgcn_rcpf(__expf(2.0f * x) + 1.0f);
}

// ---------------- grid-wide barrier (monotone counter) ----------------
__device__ __forceinline__ void grid_sync(unsigned* cnt, unsigned& epoch) {
    epoch += 1u;
    __syncthreads();
    if (threadIdx.x == 0) {
        __threadfence();                       // release prior writes (device scope)
        atomicAdd(cnt, 1u);
        unsigned tgt = epoch * (unsigned)NWG;
        while (__hip_atomic_load(cnt, __ATOMIC_RELAXED, __HIP_MEMORY_SCOPE_AGENT) < tgt)
            __builtin_amdgcn_s_sleep(2);
    }
    __syncthreads();
    __threadfence();                           // acquire
}

// ---------------- prep: pack weights to WMMA fragment layout, init state ----------------
__global__ void prep_kernel(const float* __restrict__ inputs,
                            const float* __restrict__ kern,
                            const float* __restrict__ rec,
                            const float* __restrict__ win,
                            const float* __restrict__ mix,
                            const float* __restrict__ h0,
                            const float* __restrict__ c0,
                            const float* __restrict__ w0,
                            const float* __restrict__ kappa0,
                            unsigned short* __restrict__ Wp,
                            unsigned short* __restrict__ Mp,
                            unsigned short* __restrict__ xcat,
                            float* __restrict__ cbuf,
                            float* __restrict__ kappa,
                            unsigned* __restrict__ cnt) {
    size_t tid = (size_t)blockIdx.x * blockDim.x + threadIdx.x;
    size_t nth = (size_t)gridDim.x * blockDim.x;
    if (tid == 0) *cnt = 0u;

    // big fused weight: Wp[nt(100)][kc(16)][lane(32)][j(16)] bf16
    for (size_t i = tid; i < 819200u; i += nth) {
        int j    = (int)(i & 15);
        int lane = (int)((i >> 4) & 31);
        int kc   = (int)((i >> 9) & 15);
        int nt   = (int)(i >> 13);
        int ncol = nt * 16 + (lane & 15);              // 0..1599
        int k    = kc * 32 + klocal_of(lane, j);       // 0..511 (xcat K slot)
        float v = 0.0f;
        if (k < D_IN)                          v = kern[(size_t)k * NCOLS + ncol];
        else if (k >= XOFF_H && k < XOFF_H + U_SZ) v = rec[(size_t)(k - XOFF_H) * NCOLS + ncol];
        else if (k >= XOFF_W && k < XOFF_W + W_SZ) v = win[(size_t)(k - XOFF_W) * NCOLS + ncol];
        Wp[i] = f2bf(v);
    }

    // mixture weight: Mp[nt(2)][kc(13)][lane][j] over h-slot K (0..415 -> mixture rows 0..399)
    for (size_t i = tid; i < 13312u; i += nth) {
        int j    = (int)(i & 15);
        int lane = (int)((i >> 4) & 31);
        size_t q = i >> 9;
        int kc   = (int)(q % MP_KCH);
        int nt   = (int)(q / MP_KCH);
        int ncol = nt * 16 + (lane & 15);              // 0..31 (valid < 30)
        int k    = kc * 32 + klocal_of(lane, j);       // 0..415
        float v = (k < U_SZ && ncol < 3 * K_MIX) ? mix[(size_t)k * (3 * K_MIX) + ncol] : 0.0f;
        Mp[i] = f2bf(v);
    }

    // xcat double buffer: buffer 0 = [x_0 | h0 | w0], buffer 1 = zeros
    for (size_t i = tid; i < 2u * XCAT_ELEMS; i += nth) {
        int buf = (int)(i >> 16);
        int b   = (int)((i >> 9) & 127);
        int c   = (int)(i & 511);
        float v = 0.0f;
        if (buf == 0) {
            if (c < D_IN)                           v = inputs[(size_t)b * D_IN + c];
            else if (c >= XOFF_H && c < XOFF_H + U_SZ) v = h0[(size_t)b * U_SZ + (c - XOFF_H)];
            else if (c >= XOFF_W && c < XOFF_W + W_SZ) v = w0[(size_t)b * W_SZ + (c - XOFF_W)];
        }
        xcat[i] = f2bf(v);
    }

    for (size_t i = tid; i < (size_t)B_SZ * U_SZ; i += nth) cbuf[i] = c0[i];
    for (size_t i = tid; i < (size_t)B_SZ * K_MIX; i += nth) kappa[i] = kappa0[i];
}

// ---------------- persistent recurrence kernel ----------------
__global__ __launch_bounds__(NTHREADS, 1)
void lstm_kernel(const float* __restrict__ inputs,
                 const float* __restrict__ sentence,
                 const float* __restrict__ bias,
                 const float* __restrict__ bias_mix,
                 const unsigned short* __restrict__ Wp,
                 const unsigned short* __restrict__ Mp,
                 unsigned short* __restrict__ xcat,
                 float* __restrict__ cbuf,
                 float* __restrict__ alpha,
                 float* __restrict__ beta,
                 float* __restrict__ kappa,
                 float* __restrict__ out,
                 unsigned* __restrict__ cnt) {
    extern __shared__ char smem[];                       // 64KB weights + phi scratch
    unsigned short* lw = (unsigned short*)smem;          // [g(4)][kc(16)][lane(32)][j(16)]

    const int tid  = threadIdx.x;
    const int wid  = tid >> 5;
    const int lane = tid & 31;
    const int half = lane >> 4;
    const int nn   = lane & 15;
    const int ut   = blockIdx.x;                         // unit tile 0..24
    const int bt   = wid;                                // batch tile 0..7
    const int gw   = blockIdx.x * NWAVES + wid;          // global wave id 0..199
    const int ucol = ut * 16 + nn;                       // this lane's unit column

    // -------- stage this WG's weight slice into LDS via async-to-LDS DMA --------
    #pragma unroll
    for (int g = 0; g < 4; ++g) {
        const unsigned short* src = Wp + (size_t)(g * UTILES + ut) * 8192;   // col tile g*25+ut
        unsigned short*       dst = lw + g * 8192;
        for (int i = tid; i < 1024; i += NTHREADS) {     // 1024 x 16B chunks = 16KB
            unsigned      lo = (unsigned)(uintptr_t)(dst + i * 8);
            unsigned long long ga = (unsigned long long)(uintptr_t)(src + (size_t)i * 8);
            asm volatile("global_load_async_to_lds_b128 %0, %1, off"
                         :: "v"(lo), "v"(ga) : "memory");
        }
    }
    asm volatile("s_wait_asynccnt 0" ::: "memory");
    __syncthreads();

    // per-lane gate biases (constant over time)
    const float bg0 = bias[0 * U_SZ + ucol];
    const float bg1 = bias[1 * U_SZ + ucol];
    const float bg2 = bias[2 * U_SZ + ucol];
    const float bg3 = bias[3 * U_SZ + ucol];

    unsigned epoch = 0u;

    for (int t = 0; t < T_STEPS; ++t) {
        const unsigned short* xc = xcat + (size_t)(t & 1) * XCAT_ELEMS;        // holds x_t,h_{t-1},w_{t-1}
        unsigned short*       xn = xcat + (size_t)((t + 1) & 1) * XCAT_ELEMS;  // receives h_t,w_t,x_{t+1}

        // ================= phase A: fused gate GEMM + LSTM cell =================
        {
            v8f ai = {}, af_ = {}, ag = {}, ao = {};
            #pragma unroll 4
            for (int kc = 0; kc < KCH; ++kc) {
                Frag fa;
                const unsigned short* rp = xc + (size_t)(bt * 16 + nn) * KPAD + kc * 32 + half * 8;
                fa.u[0] = *(const v4u*)(rp);
                fa.u[1] = *(const v4u*)(rp + 16);

                const unsigned short* lp = lw + ((size_t)kc * 32 + lane) * 16;
                Frag fb;
                fb.u[0] = *(const v4u*)(lp);
                fb.u[1] = *(const v4u*)(lp + 8);
                ai = __builtin_amdgcn_wmma_f32_16x16x32_bf16(false, fa.h, false, fb.h, (short)0, ai, false, false);
                fb.u[0] = *(const v4u*)(lp + 8192);
                fb.u[1] = *(const v4u*)(lp + 8200);
                af_ = __builtin_amdgcn_wmma_f32_16x16x32_bf16(false, fa.h, false, fb.h, (short)0, af_, false, false);
                fb.u[0] = *(const v4u*)(lp + 16384);
                fb.u[1] = *(const v4u*)(lp + 16392);
                ag = __builtin_amdgcn_wmma_f32_16x16x32_bf16(false, fa.h, false, fb.h, (short)0, ag, false, false);
                fb.u[0] = *(const v4u*)(lp + 24576);
                fb.u[1] = *(const v4u*)(lp + 24584);
                ao = __builtin_amdgcn_wmma_f32_16x16x32_bf16(false, fa.h, false, fb.h, (short)0, ao, false, false);
            }
            // gate fusion on the 16x16 C tile this wave owns
            #pragma unroll
            for (int r = 0; r < 8; ++r) {
                int b = bt * 16 + r + half * 8;          // C/D layout: lanes>=16 hold M=r+8
                float ig = sigmoidf_(ai[r] + bg0);
                float fg = sigmoidf_(af_[r] + bg1);
                float gg = tanhf_(ag[r] + bg2);
                float og = sigmoidf_(ao[r] + bg3);
                float cprev = cbuf[(size_t)b * U_SZ + ucol];
                float cnew  = fg * cprev + ig * gg;
                float hnew  = og * tanhf_(cnew);
                cbuf[(size_t)b * U_SZ + ucol] = cnew;
                out[((size_t)t * B_SZ + b) * OUTC + ucol] = hnew;
                xn[(size_t)b * KPAD + XOFF_H + ucol] = f2bf(hnew);
            }
        }
        grid_sync(cnt, epoch);

        // ================= phase B1: mixture GEMM -> alpha/beta/kappa =================
        if (gw < 16) {
            int nt  = gw & 1;
            int btm = gw >> 1;
            v8f ma = {};
            #pragma unroll
            for (int kc = 0; kc < MP_KCH; ++kc) {
                Frag fa;
                const unsigned short* rp = xn + (size_t)(btm * 16 + nn) * KPAD + XOFF_H + kc * 32 + half * 8;
                fa.u[0] = *(const v4u*)(rp);
                fa.u[1] = *(const v4u*)(rp + 16);
                Frag fb;
                const unsigned short* mp = Mp + ((size_t)(nt * MP_KCH + kc) * 32 + lane) * 16;
                fb.u[0] = *(const v4u*)(mp);
                fb.u[1] = *(const v4u*)(mp + 8);
                ma = __builtin_amdgcn_wmma_f32_16x16x32_bf16(false, fa.h, false, fb.h, (short)0, ma, false, false);
            }
            int col = nt * 16 + nn;                      // 0..31, valid < 30
            if (col < 3 * K_MIX) {
                float bm = bias_mix[col];
                #pragma unroll
                for (int r = 0; r < 8; ++r) {
                    int b = btm * 16 + r + half * 8;
                    float e = __expf(ma[r] + bm);
                    if (col < K_MIX)            alpha[(size_t)b * K_MIX + col] = e;
                    else if (col < 2 * K_MIX)   beta[(size_t)b * K_MIX + (col - K_MIX)] = e;
                    else {
                        int kk = col - 2 * K_MIX;
                        float kp = kappa[(size_t)b * K_MIX + kk] + e;
                        kappa[(size_t)b * K_MIX + kk] = kp;
                    }
                }
            }
        }
        grid_sync(cnt, epoch);

        // ================= phase B2: phi + window einsum + stage next inputs =================
        if (gw < B_SZ) {
            int b = gw;
            float* phis = (float*)(smem + 65536) + wid * L_SZ;
            float al[K_MIX], be[K_MIX], ka[K_MIX];
            #pragma unroll
            for (int k = 0; k < K_MIX; ++k) {
                al[k] = alpha[(size_t)b * K_MIX + k];
                be[k] = beta[(size_t)b * K_MIX + k];
                ka[k] = kappa[(size_t)b * K_MIX + k];
            }
            #pragma unroll
            for (int ll = 0; ll < 2; ++ll) {
                int l = lane + 32 * ll;
                float uu = (float)(l + 1);
                float s = 0.0f;
                #pragma unroll
                for (int k = 0; k < K_MIX; ++k) {
                    float d = ka[k] - uu;
                    s += al[k] * __expf(-be[k] * d * d);
                }
                phis[l] = s;
            }
            asm volatile("s_wait_dscnt 0" ::: "memory");   // wave-synchronous phi exchange via LDS
            for (int j = lane; j < W_SZ; j += 32) {
                const float* sp = sentence + (size_t)b * L_SZ * W_SZ + j;
                float acc = 0.0f;
                #pragma unroll 8
                for (int l = 0; l < L_SZ; ++l) acc += phis[l] * sp[(size_t)l * W_SZ];
                out[((size_t)t * B_SZ + b) * OUTC + U_SZ + j] = acc;
                xn[(size_t)b * KPAD + XOFF_W + j] = f2bf(acc);
            }
            if (lane < D_IN && (t + 1) < T_STEPS)
                xn[(size_t)b * KPAD + lane] =
                    f2bf(inputs[((size_t)(t + 1) * B_SZ + b) * D_IN + lane]);
        }
        grid_sync(cnt, epoch);
    }
}

// ---------------- launch ----------------
extern "C" void kernel_launch(void* const* d_in, const int* in_sizes, int n_in,
                              void* d_out, int out_size, void* d_ws, size_t ws_size,
                              hipStream_t stream) {
    const float* inputs   = (const float*)d_in[0];
    const float* sentence = (const float*)d_in[1];
    const float* kern     = (const float*)d_in[2];
    const float* rec      = (const float*)d_in[3];
    const float* win      = (const float*)d_in[4];
    const float* mix      = (const float*)d_in[5];
    const float* bias     = (const float*)d_in[6];
    const float* bias_mix = (const float*)d_in[7];
    const float* h0       = (const float*)d_in[8];
    const float* c0       = (const float*)d_in[9];
    const float* w0       = (const float*)d_in[10];
    const float* kappa0   = (const float*)d_in[11];

    char* ws = (char*)d_ws;
    unsigned*       cnt   = (unsigned*)(ws + WS_CNT);
    unsigned short* Wp    = (unsigned short*)(ws + WS_WP);
    unsigned short* Mp    = (unsigned short*)(ws + WS_MP);
    unsigned short* xcat  = (unsigned short*)(ws + WS_XCAT);
    float*          cbuf  = (float*)(ws + WS_C);
    float*          alpha = (float*)(ws + WS_ALPHA);
    float*          beta  = (float*)(ws + WS_BETA);
    float*          kap   = (float*)(ws + WS_KAPPA);
    float*          out   = (float*)d_out;

    prep_kernel<<<512, 256, 0, stream>>>(inputs, kern, rec, win, mix, h0, c0, w0,
                                         kappa0, Wp, Mp, xcat, cbuf, kap, cnt);

    const size_t lds_bytes = 65536 + NWAVES * L_SZ * sizeof(float);   // weights + phi scratch
    lstm_kernel<<<NWG, NTHREADS, lds_bytes, stream>>>(inputs, sentence, bias, bias_mix,
                                                      Wp, Mp, xcat, cbuf, alpha, beta,
                                                      kap, out, cnt);
}
